// CRF_19086834663558
// MI455X (gfx1250) — matrix operand order repaired and checked by reading, not compiled, not run on hardware
//
#include <hip/hip_runtime.h>

// ---------------------------------------------------------------------------
// CRF log-partition on MI455X (gfx1250), chunked matrix-product scan.
//
//   logZ = logz0 + log( beta0^T * PROD_{t=1..L-1} (Tm * diag(ex_t)) * 1 )
//
//  K1 crf_tmt    : Tm -> TmT bf16 (B-fragment friendly, [n][k])
//  K2 crf_ex     : gather ex_t[j] = E[j, x[t]]            (L x 512 f32)
//  K3 crf_prep   : phi0 -> beta0, logz0 ; path score
//  K4 crf_chunk  : 32 blocks x 1024 thr; 128 bf16 WMMA GEMM steps per chunk,
//                  register-blocked 64x32 per wave-group (8 WMMA / 12 loads),
//                  lazy max-normalization with per-chunk log-scale,
//                  async global->LDS staging of the ex row (ASYNCcnt path)
//  K5 crf_combine: 1 block; beta0 through 32 chunk matrices, final scalar
// ---------------------------------------------------------------------------

#define M_TAGS   512
#define V_VOCAB  50000
#define L_SEQ    4096
#define NCHUNK   32
#define KSTEP    128                 // steps per chunk (last chunk: 127)
#define MATELEMS (512 * 512)         // elements per 512x512 matrix

typedef __attribute__((ext_vector_type(16))) __bf16 v16bf;
typedef __attribute__((ext_vector_type(8)))  float  v8f;

#if defined(__gfx1250__) && __has_builtin(__builtin_amdgcn_global_load_async_to_lds_b32) && __has_builtin(__builtin_amdgcn_s_wait_asynccnt)
#define HAVE_ASYNC_LDS 1
typedef __attribute__((address_space(1))) int GINT;   // global (AS1) int
typedef __attribute__((address_space(3))) int LINT;   // LDS    (AS3) int
#else
#define HAVE_ASYNC_LDS 0
#endif

static __device__ __forceinline__ unsigned short f2bf(float f) {
    unsigned int u = __float_as_uint(f);                 // round-to-nearest-even
    unsigned int r = (u + 0x7FFFu + ((u >> 16) & 1u)) >> 16;
    return (unsigned short)r;
}
static __device__ __forceinline__ float bf2f(unsigned short h) {
    return __uint_as_float(((unsigned int)h) << 16);
}
static __device__ __forceinline__ v16bf frag_ld(const unsigned short* p, int hi_elems) {
    union { uint4 u[2]; v16bf v; } x;
    x.u[0] = *(const uint4*)(p);
    x.u[1] = *(const uint4*)(p + hi_elems);
    return x.v;
}

// ---------------- K1: TmT[n][k] = bf16(T[k][n]) -----------------------------
__global__ void crf_tmt_kernel(const float* __restrict__ T,
                               unsigned short* __restrict__ TmT) {
    int n = blockIdx.x, k = threadIdx.x;                 // 512 x 512
    TmT[(size_t)n * 512 + k] = f2bf(T[(size_t)k * 512 + n]);
}

// ---------------- K2: Ex[t][j] = E[j][x[t]] --------------------------------
__global__ void crf_ex_kernel(const float* __restrict__ E,
                              const int* __restrict__ x,
                              float* __restrict__ Ex) {
    int t = blockIdx.x, j = threadIdx.x;                 // L_SEQ x 512
    Ex[(size_t)t * 512 + j] = E[(size_t)j * V_VOCAB + x[t]];
}

// ---------------- K3: phi0 / beta0 / logz0 and path score ------------------
__global__ void crf_prep_kernel(const float* __restrict__ T,
                                const float* __restrict__ E,
                                const float* __restrict__ Eprev,
                                const float* __restrict__ Enext,
                                const float* __restrict__ Cap,
                                const int* __restrict__ x,
                                const int* __restrict__ y,
                                const int* __restrict__ up,
                                float* __restrict__ beta0,
                                float* __restrict__ scal) {  // scal[0]=logz0 scal[1]=pathsum
    __shared__ float red[512];
    const int j = threadIdx.x;
    const int VP = V_VOCAB + 1;
    int x0 = x[0], x1 = x[1], u0 = up[0];
    float phi = T[(size_t)M_TAGS * 512 + j]
              + Eprev[(size_t)j * VP + M_TAGS]
              + Enext[(size_t)j * VP + x1]
              + Cap[(size_t)j * 2 + u0]
              + E[(size_t)j * V_VOCAB + x0];
    float a = __expf(phi);
    red[j] = a; __syncthreads();
    for (int off = 256; off >= 1; off >>= 1) {
        if (j < off) red[j] += red[j + off];
        __syncthreads();
    }
    float s = red[0];
    beta0[j] = a / s;
    if (j == 0) scal[0] = __logf(s);
    __syncthreads();                                     // all reads of red[0] done
    float acc = 0.f;
    for (int t = j; t < L_SEQ; t += 512) {
        int yt = y[t];
        int yp = (t == 0) ? M_TAGS : y[t - 1];
        int xp = (t == 0) ? M_TAGS : x[t - 1];
        int xn = (t == L_SEQ - 1) ? M_TAGS : x[t + 1];
        acc += T[(size_t)yp * 512 + yt]
             + Eprev[(size_t)yt * VP + xp]
             + Enext[(size_t)yt * VP + xn]
             + Cap[(size_t)yt * 2 + up[t]]
             + E[(size_t)yt * V_VOCAB + x[t]];
    }
    red[j] = acc; __syncthreads();
    for (int off = 256; off >= 1; off >>= 1) {
        if (j < off) red[j] += red[j + off];
        __syncthreads();
    }
    if (j == 0) scal[1] = red[0];
}

// ---------------- K4: chunk partial products via WMMA ----------------------
// One block per chunk; 32 waves. 512x512 output split into 128 wave-groups of
// 64 rows x 32 cols (4x2 tiles of 16x16); each wave owns 4 groups.
__global__ void __launch_bounds__(1024)
crf_chunk_kernel(const unsigned short* __restrict__ TmT,
                 const float* __restrict__ Ex,
                 unsigned short* __restrict__ bufBase,
                 float* __restrict__ logAcc) {
    const int c    = blockIdx.x;
    const int tid  = threadIdx.x;
    const int lane = tid & 31;
    const int wave = tid >> 5;

    unsigned short* buf0 = bufBase + (size_t)c * (2 * MATELEMS);
    unsigned short* buf1 = buf0 + MATELEMS;

    __shared__ float s_ex[512];
    __shared__ float s_wmax[32];
    __shared__ float s_m;
    __shared__ float s_logAcc;

    // A := I (bf16 identity) in buf0
    for (int idx = tid; idx < MATELEMS; idx += 1024) {
        int r = idx >> 9, col = idx & 511;
        buf0[idx] = (r == col) ? (unsigned short)0x3F80u : (unsigned short)0u;
    }
    if (tid == 0) { s_m = 1.0f; s_logAcc = 0.0f; }

    const int tBegin = 1 + c * KSTEP;
    int tEnd = tBegin + KSTEP; if (tEnd > L_SEQ) tEnd = L_SEQ;

    unsigned short* rd = buf0;
    unsigned short* wr = buf1;

    const int sub  = lane & 15;
    const int hiA  = (lane < 16) ? 0 : 8;    // A-fragment K sub-offset (elements)
    const int hiB  = (lane < 16) ? 0 : 8;    // B-fragment K sub-offset (elements)
    const int rofs = (lane < 16) ? 0 : 8;    // C/D row sub-offset

    for (int t = tBegin; t < tEnd; ++t) {
#if HAVE_ASYNC_LDS
        if (tid < 512) {
            __builtin_amdgcn_global_load_async_to_lds_b32(
                (GINT*)(Ex + (size_t)t * 512 + tid), (LINT*)&s_ex[tid], 0, 0);
            __builtin_amdgcn_s_wait_asynccnt(0);
        }
#else
        if (tid < 512) s_ex[tid] = Ex[(size_t)t * 512 + tid];
#endif
        __syncthreads();                                  // ex + s_m + rd ready
        const float invm = 1.0f / s_m;
        float lmax = 0.0f;

        for (int g = wave; g < 128; g += 32) {
            const int r0 = (g >> 4) << 6;                 // 64-row band
            const int n0 = (g & 15) << 5;                 // 32-col band
            v8f acc[4][2];
            #pragma unroll
            for (int i = 0; i < 4; ++i)
                #pragma unroll
                for (int q = 0; q < 2; ++q) acc[i][q] = (v8f){};

            const unsigned short* abase = rd  + (size_t)(r0 + sub) * 512 + hiA;
            const unsigned short* bbase = TmT + (size_t)(n0 + sub) * 512 + hiB;

            #pragma unroll 2
            for (int kb = 0; kb < 16; ++kb) {
                const int k0 = kb << 5;
                v16bf af[4], bf[2];
                #pragma unroll
                for (int i = 0; i < 4; ++i)
                    af[i] = frag_ld(abase + (size_t)(i * 16) * 512 + k0, 16);
                #pragma unroll
                for (int q = 0; q < 2; ++q)
                    bf[q] = frag_ld(bbase + (size_t)(q * 16) * 512 + k0, 16);
                #pragma unroll
                for (int i = 0; i < 4; ++i)
                    #pragma unroll
                    for (int q = 0; q < 2; ++q)
                        acc[i][q] = __builtin_amdgcn_wmma_f32_16x16x32_bf16(
                                        false, af[i], false, bf[q],
                                        (short)0, acc[i][q], false, false);
            }
            #pragma unroll
            for (int q = 0; q < 2; ++q) {
                const int ncol = n0 + q * 16 + sub;
                const float ecol = s_ex[ncol] * invm;     // column scale, lazy 1/m
                #pragma unroll
                for (int i = 0; i < 4; ++i) {
                    const int rbase = r0 + i * 16 + rofs;
                    #pragma unroll
                    for (int j = 0; j < 8; ++j) {
                        float v = acc[i][q][j] * ecol;
                        lmax = fmaxf(lmax, v);
                        wr[(size_t)(rbase + j) * 512 + ncol] = f2bf(v);
                    }
                }
            }
        }

        for (int off = 16; off >= 1; off >>= 1)
            lmax = fmaxf(lmax, __shfl_xor(lmax, off, 32));
        if (lane == 0) s_wmax[wave] = lmax;
        __syncthreads();                                  // tiles written, maxes ready
        if (wave == 0) {
            float m = s_wmax[lane];
            for (int off = 16; off >= 1; off >>= 1)
                m = fmaxf(m, __shfl_xor(m, off, 32));
            if (lane == 0) {
                s_logAcc += __logf(s_m);                  // scale applied this step
                s_m = fmaxf(m, 1e-30f);
            }
        }
        unsigned short* tmp = rd; rd = wr; wr = tmp;      // next iter's barrier orders s_m
    }
    if (tid == 0) logAcc[c] = s_logAcc;
}

// ---------------- K5: serial combine -----------------------------------------
__global__ void crf_combine_kernel(const unsigned short* __restrict__ bufBase,
                                   const float* __restrict__ logAcc,
                                   const float* __restrict__ beta0,
                                   const float* __restrict__ scal,
                                   float* __restrict__ out) {
    __shared__ float s_r[512];
    __shared__ float s_red[512];
    __shared__ float s_logz;
    const int j = threadIdx.x;
    s_r[j] = beta0[j];
    if (j == 0) s_logz = scal[0];
    __syncthreads();
    for (int c = 0; c < NCHUNK; ++c) {
        int nsteps = (c < NCHUNK - 1) ? KSTEP : (KSTEP - 1);
        const unsigned short* P = bufBase + (size_t)c * (2 * MATELEMS)
                                + ((nsteps & 1) ? MATELEMS : 0);
        float v = 0.f;
        for (int k = 0; k < 512; ++k)                     // coalesced over j
            v += s_r[k] * bf2f(P[(size_t)k * 512 + j]);
        s_red[j] = v; __syncthreads();
        for (int off = 256; off >= 1; off >>= 1) {
            if (j < off) s_red[j] += s_red[j + off];
            __syncthreads();
        }
        float s = s_red[0];
        s_r[j] = v / s;
        if (j == 0) s_logz += __logf(s) + logAcc[c];
        __syncthreads();
    }
    if (j == 0) out[0] = s_logz - scal[1];
}

// ---------------------------------------------------------------------------
extern "C" void kernel_launch(void* const* d_in, const int* in_sizes, int n_in,
                              void* d_out, int out_size, void* d_ws, size_t ws_size,
                              hipStream_t stream) {
    const float* T     = (const float*)d_in[0];
    const float* E     = (const float*)d_in[1];
    const float* Eprev = (const float*)d_in[2];
    const float* Enext = (const float*)d_in[3];
    const float* Cap   = (const float*)d_in[4];
    const int*   x     = (const int*)d_in[5];
    const int*   y     = (const int*)d_in[6];
    const int*   up    = (const int*)d_in[7];
    float* out = (float*)d_out;

    char* ws = (char*)d_ws;
    unsigned short* TmT   = (unsigned short*)(ws);                 // 512 KB
    float*          Ex    = (float*)(ws + 524288);                 // 8 MB
    float*          lAcc  = (float*)(ws + 8912896);                // 128 B
    float*          beta0 = (float*)(ws + 8913024);                // 2 KB
    float*          scal  = (float*)(ws + 8915072);                // 8 B
    unsigned short* bufs  = (unsigned short*)(ws + 8916992);       // 32 MB ping-pong

    crf_tmt_kernel<<<dim3(512), dim3(512), 0, stream>>>(T, TmT);
    crf_ex_kernel<<<dim3(L_SEQ), dim3(512), 0, stream>>>(E, x, Ex);
    crf_prep_kernel<<<dim3(1), dim3(512), 0, stream>>>(T, E, Eprev, Enext, Cap,
                                                       x, y, up, beta0, scal);
    crf_chunk_kernel<<<dim3(NCHUNK), dim3(1024), 0, stream>>>(TmT, Ex, bufs, lAcc);
    crf_combine_kernel<<<dim3(1), dim3(512), 0, stream>>>(bufs, lAcc, beta0, scal, out);
    (void)in_sizes; (void)n_in; (void)out_size; (void)ws_size;
}
